// Mamba2Mixin_5188320493604
// MI455X (gfx1250) — compile-verified, matrix operand order checked by reading, not run on hardware
//
#include <hip/hip_runtime.h>
#include <hip/hip_bf16.h>

// ---------------- Mamba2 fixed config ----------------
#define BATCH   4
#define SEQ     2048
#define ROWS    (BATCH * SEQ)        // 8192 token rows
#define DM      2048                 // d_model = H*P
#define HH      32                   // heads
#define PP      64                   // head dim
#define NN      32                   // state size
#define KK      4                    // conv kernel
#define CONV    (DM + 2 * NN)        // 2112
#define PROJ    (DM + CONV + HH)     // 4192
#define PROJP   4224                 // PROJ padded to multiple of 128
#define EPSV    1e-6f

typedef __attribute__((ext_vector_type(16))) __bf16 v16bf;
typedef __attribute__((ext_vector_type(8)))  float  v8f;

// round-to-nearest-even f32 -> bf16 (as u16 bits)
__device__ __forceinline__ unsigned short f2bf(float f) {
    unsigned int u = __float_as_uint(f);
    u += 0x7FFFu + ((u >> 16) & 1u);
    return (unsigned short)(u >> 16);
}
__device__ __forceinline__ float sigmoidf_(float x) { return 1.0f / (1.0f + __expf(-x)); }

// LDS byte address for async-to-LDS VDST: flat shared pointer keeps the LDS
// byte offset in addr[31:0] (LDS aperture), so truncation is the raw DS address.
__device__ __forceinline__ unsigned lds_addr_of(const void* p) {
    return (unsigned)(unsigned long long)(uintptr_t)p;
}

// one per-lane 16B async move: LDS[vdst] = MEM[sbase + voff]  (ASYNCcnt)
__device__ __forceinline__ void async_ld_b128(unsigned lds_addr, unsigned long long gbase, unsigned voff) {
    asm volatile("global_load_async_to_lds_b128 %0, %1, %2"
                 :: "v"(lds_addr), "v"(voff), "s"(gbase)
                 : "memory");
}
__device__ __forceinline__ void wait_asynccnt0() {
#if __has_builtin(__builtin_amdgcn_s_wait_asynccnt)
    __builtin_amdgcn_s_wait_asynccnt(0);
#else
    asm volatile("s_wait_asynccnt 0x0" ::: "memory");
#endif
}

// ---------------- conversion / padding kernels ----------------
__global__ void k_f32_to_bf16(const float* __restrict__ in, unsigned short* __restrict__ out, int n) {
    int i = blockIdx.x * 256 + threadIdx.x;
    if (i < n) out[i] = f2bf(in[i]);
}

// W_in (K=2048 x N=4192) -> transposed + padded bf16 Wt[N=4224][K=2048]
__global__ void k_cvt_WinT(const float* __restrict__ in, unsigned short* __restrict__ out) {
    int i = blockIdx.x * 256 + threadIdx.x;          // over PROJP * DM
    if (i >= PROJP * DM) return;
    int n = i / DM, k = i - n * DM;
    float v = (n < PROJ) ? in[(size_t)k * PROJ + n] : 0.0f;
    out[i] = f2bf(v);
}

// W_out (K=2048 x N=2048) -> transposed bf16 Wt[N][K]
__global__ void k_cvt_WoutT(const float* __restrict__ in, unsigned short* __restrict__ out) {
    int i = blockIdx.x * 256 + threadIdx.x;          // over DM * DM
    if (i >= DM * DM) return;
    int n = i / DM, k = i - n * DM;
    out[i] = f2bf(in[(size_t)k * DM + n]);
}

__global__ void k_pad_bias(const float* __restrict__ in, float* __restrict__ out) {
    int i = blockIdx.x * 256 + threadIdx.x;
    if (i < PROJP) out[i] = (i < PROJ) ? in[i] : 0.0f;
}

// ---------------- bf16 WMMA GEMM: C[M,N] = A[M,K] * BT[N,K]^T (+bias) ----------------
// block = 128 threads (4 waves), tile 128x128, k-step 32, double-buffered LDS
// filled by global_load_async_to_lds_b128 (ASYNCcnt). Wave w computes a 64x64
// subtile via 4x4 v_wmma_f32_16x16x32_bf16.
#define LDSTRIDE 40                       // halves; 80B rows: 16B aligned, bank-staggered
#define TILEHALF (128 * LDSTRIDE)         // halves per tile buffer
#define TILEBYTE (TILEHALF * 2)           // 10240 B

__global__ void __launch_bounds__(128)
k_gemm_bf16(const unsigned short* __restrict__ A,    // [Mdim][Kdim]
            const unsigned short* __restrict__ BT,   // [Ndim][Kdim]
            const float* __restrict__ bias,
            float* __restrict__ C,
            int Mdim, int Ndim, int Kdim) {
    __shared__ unsigned short Asm[2][TILEHALF];
    __shared__ unsigned short Bsm[2][TILEHALF];
    (void)Mdim;

    const int tid  = threadIdx.x;
    const int lane = tid & 31;
    const int wave = tid >> 5;
    const int m0 = blockIdx.y * 128;
    const int n0 = blockIdx.x * 128;
    const int wm = (wave >> 1) * 64;
    const int wn = (wave & 1) * 64;
    const int half = lane >> 4;
    const int lrow = lane & 15;

    const unsigned long long Abase = (unsigned long long)(uintptr_t)A;
    const unsigned long long Bbase = (unsigned long long)(uintptr_t)BT;
    const unsigned aRowOff = (unsigned)(m0 + tid) * (unsigned)Kdim * 2u;  // bytes
    const unsigned bRowOff = (unsigned)(n0 + tid) * (unsigned)Kdim * 2u;  // bytes
    const unsigned ldsA0 = lds_addr_of(&Asm[0][tid * LDSTRIDE]);
    const unsigned ldsB0 = lds_addr_of(&Bsm[0][tid * LDSTRIDE]);

    // issue one 128x32 A tile + 128x32 B tile into buffer `buf` (8 async b128/thread)
    auto prefetch = [&](int buf, int k0) {
        const unsigned kb = (unsigned)k0 * 2u;
#pragma unroll
        for (int j = 0; j < 4; ++j) {
            async_ld_b128(ldsA0 + (unsigned)buf * TILEBYTE + j * 16, Abase, aRowOff + kb + j * 16);
            async_ld_b128(ldsB0 + (unsigned)buf * TILEBYTE + j * 16, Bbase, bRowOff + kb + j * 16);
        }
    };

    v8f acc[4][4];
#pragma unroll
    for (int mi = 0; mi < 4; ++mi)
#pragma unroll
        for (int ni = 0; ni < 4; ++ni) acc[mi][ni] = (v8f)0.0f;

    const int nk = Kdim >> 5;
    prefetch(0, 0);

    for (int kt = 0; kt < nk; ++kt) {
        const int cur = kt & 1;
        wait_asynccnt0();          // this wave's tile data for `cur` has landed
        __syncthreads();           // everyone's data landed; everyone done reading cur^1
        if (kt + 1 < nk) prefetch(cur ^ 1, (kt + 1) << 5);

        // fragments (ISA 16-bit layout: lanes0-15 K=0..7,16..23; lanes16-31 K=8..15,24..31)
        v16bf af[4], bf[4];
#pragma unroll
        for (int mi = 0; mi < 4; ++mi) {
            const unsigned short* p = &Asm[cur][(wm + mi * 16 + lrow) * LDSTRIDE + half * 8];
            ((uint4*)&af[mi])[0] = *(const uint4*)p;
            ((uint4*)&af[mi])[1] = *(const uint4*)(p + 16);
        }
#pragma unroll
        for (int ni = 0; ni < 4; ++ni) {
            const unsigned short* p = &Bsm[cur][(wn + ni * 16 + lrow) * LDSTRIDE + half * 8];
            ((uint4*)&bf[ni])[0] = *(const uint4*)p;
            ((uint4*)&bf[ni])[1] = *(const uint4*)(p + 16);
        }
#pragma unroll
        for (int mi = 0; mi < 4; ++mi)
#pragma unroll
            for (int ni = 0; ni < 4; ++ni)
                acc[mi][ni] = __builtin_amdgcn_wmma_f32_16x16x32_bf16(
                    false, af[mi], false, bf[ni], (short)0, acc[mi][ni], false, false);
    }

    // epilogue: D layout VGPR r -> row r + 8*half, col = lane&15
#pragma unroll
    for (int mi = 0; mi < 4; ++mi)
#pragma unroll
        for (int ni = 0; ni < 4; ++ni) {
            const int col = n0 + wn + ni * 16 + lrow;
            const float bv = bias ? bias[col] : 0.0f;
#pragma unroll
            for (int r = 0; r < 8; ++r) {
                const int row = m0 + wm + mi * 16 + half * 8 + r;
                C[(size_t)row * Ndim + col] = acc[mi][ni][r] + bv;
            }
        }
}

// ---------------- causal depthwise conv1d + SiLU ----------------
__global__ void k_conv_silu(const float* __restrict__ proj,
                            const float* __restrict__ cw,
                            const float* __restrict__ cb,
                            float* __restrict__ out) {
    int i = blockIdx.x * 256 + threadIdx.x;      // over ROWS*CONV
    if (i >= ROWS * CONV) return;
    int row = i / CONV, c = i - row * CONV;
    int l = row & (SEQ - 1);
    float acc = cb[c];
#pragma unroll
    for (int k = 0; k < KK; ++k) {
        int dl = l - (KK - 1) + k;
        if (dl >= 0)
            acc = fmaf(proj[(size_t)(row - (KK - 1) + k) * PROJP + DM + c], cw[c * KK + k], acc);
    }
    out[i] = acc * sigmoidf_(acc);               // SiLU
}

// ---------------- dt = softplus(dt_raw + bias); dA = exp(dt * -exp(A_log)) ----------------
__global__ void k_dt(const float* __restrict__ proj,
                     const float* __restrict__ dt_bias,
                     const float* __restrict__ A_log,
                     float* __restrict__ dt, float* __restrict__ dA) {
    int i = blockIdx.x * 256 + threadIdx.x;      // over ROWS*HH
    if (i >= ROWS * HH) return;
    int row = i / HH, h = i - row * HH;
    float s = proj[(size_t)row * PROJP + (DM + CONV) + h] + dt_bias[h];
    float d = (s > 20.0f) ? s : log1pf(__expf(s));
    float a = -__expf(A_log[h]);
    dt[i] = d;
    dA[i] = __expf(d * a);
}

// ---------------- selective scan ----------------
__global__ void __launch_bounds__(64)
k_scan(const float* __restrict__ convout,
       const float* __restrict__ dt, const float* __restrict__ dA,
       const float* __restrict__ Dp, float* __restrict__ y) {
    const int b = blockIdx.x / HH;
    const int h = blockIdx.x - b * HH;
    const int p = threadIdx.x;
    __shared__ float sB[NN];
    __shared__ float sC[NN];
    float hs[NN];
#pragma unroll
    for (int n = 0; n < NN; ++n) hs[n] = 0.0f;
    const float Dh = Dp[h];

    for (int l = 0; l < SEQ; ++l) {
        const size_t row = (size_t)b * SEQ + l;
        if (p < NN)            sB[p]      = convout[row * CONV + DM + p];
        else                   sC[p - NN] = convout[row * CONV + DM + NN + (p - NN)];
        __syncthreads();
        const float dA_t = dA[row * HH + h];
        const float dt_t = dt[row * HH + h];
        const float x_t  = convout[row * CONV + h * PP + p];
        const float coef = dt_t * x_t;
        float acc = 0.0f;
#pragma unroll
        for (int n = 0; n < NN; ++n) {
            hs[n] = fmaf(dA_t, hs[n], coef * sB[n]);
            acc = fmaf(hs[n], sC[n], acc);
        }
        y[row * DM + h * PP + p] = acc + Dh * x_t;
        __syncthreads();
    }
}

// ---------------- gated RMSNorm -> bf16 ----------------
__global__ void __launch_bounds__(256)
k_rmsnorm(const float* __restrict__ proj,       // gate = cols [0, DM)
          const float* __restrict__ y,
          const float* __restrict__ nw,
          unsigned short* __restrict__ out_bf) {
    __shared__ float red[256];
    const int row = blockIdx.x;
    const int tid = threadIdx.x;
    float t8[DM / 256];
    float ss = 0.0f;
#pragma unroll
    for (int i = 0; i < DM / 256; ++i) {
        const int c = i * 256 + tid;
        const float g = proj[(size_t)row * PROJP + c];
        const float t = y[(size_t)row * DM + c] * (g * sigmoidf_(g));
        t8[i] = t;
        ss = fmaf(t, t, ss);
    }
    red[tid] = ss;
    __syncthreads();
    for (int s = 128; s > 0; s >>= 1) {
        if (tid < s) red[tid] += red[tid + s];
        __syncthreads();
    }
    const float scale = rsqrtf(red[0] * (1.0f / (float)DM) + EPSV);
#pragma unroll
    for (int i = 0; i < DM / 256; ++i) {
        const int c = i * 256 + tid;
        out_bf[(size_t)row * DM + c] = f2bf(t8[i] * scale * nw[c]);
    }
}

// ---------------- launcher ----------------
extern "C" void kernel_launch(void* const* d_in, const int* in_sizes, int n_in,
                              void* d_out, int out_size, void* d_ws, size_t ws_size,
                              hipStream_t stream) {
    const float* x       = (const float*)d_in[0];
    const float* W_in    = (const float*)d_in[1];
    const float* b_in    = (const float*)d_in[2];
    const float* conv_w  = (const float*)d_in[3];
    const float* conv_b  = (const float*)d_in[4];
    const float* A_log   = (const float*)d_in[5];
    const float* dt_bias = (const float*)d_in[6];
    const float* D_param = (const float*)d_in[7];
    const float* norm_w  = (const float*)d_in[8];
    const float* W_out   = (const float*)d_in[9];
    float* out = (float*)d_out;

    char* w = (char*)d_ws;
    size_t off = 0;
    auto alloc = [&](size_t bytes) -> void* {
        void* p = w + off;
        off = (off + bytes + 255) & ~(size_t)255;
        return p;
    };
    unsigned short* x_bf     = (unsigned short*)alloc((size_t)ROWS * DM * 2);
    unsigned short* WinT_bf  = (unsigned short*)alloc((size_t)PROJP * DM * 2);
    unsigned short* WoutT_bf = (unsigned short*)alloc((size_t)DM * DM * 2);
    float*          biasp    = (float*)alloc((size_t)PROJP * 4);
    float*          proj     = (float*)alloc((size_t)ROWS * PROJP * 4);
    float*          convout  = (float*)alloc((size_t)ROWS * CONV * 4);
    float*          dtb      = (float*)alloc((size_t)ROWS * HH * 4);
    float*          dAb      = (float*)alloc((size_t)ROWS * HH * 4);
    float*          ybuf     = (float*)alloc((size_t)ROWS * DM * 4);
    unsigned short* yn_bf    = (unsigned short*)alloc((size_t)ROWS * DM * 2);
    (void)ws_size; (void)n_in; (void)in_sizes; (void)out_size;

    // conversions (weights transposed to [N][K] so GEMM stages both tiles contiguously)
    {
        int n = ROWS * DM;
        k_f32_to_bf16<<<(n + 255) / 256, 256, 0, stream>>>(x, x_bf, n);
    }
    k_cvt_WinT<<<(PROJP * DM + 255) / 256, 256, 0, stream>>>(W_in, WinT_bf);
    k_cvt_WoutT<<<(DM * DM + 255) / 256, 256, 0, stream>>>(W_out, WoutT_bf);
    k_pad_bias<<<(PROJP + 255) / 256, 256, 0, stream>>>(b_in, biasp);

    // GEMM1: proj = x @ W_in + b_in   (8192 x 4224 x 2048)
    {
        dim3 grid(PROJP / 128, ROWS / 128);
        k_gemm_bf16<<<grid, 128, 0, stream>>>(x_bf, WinT_bf, biasp, proj, ROWS, PROJP, DM);
    }

    // conv + SiLU
    k_conv_silu<<<(ROWS * CONV + 255) / 256, 256, 0, stream>>>(proj, conv_w, conv_b, convout);

    // dt / dA
    k_dt<<<(ROWS * HH + 255) / 256, 256, 0, stream>>>(proj, dt_bias, A_log, dtb, dAb);

    // selective scan
    k_scan<<<BATCH * HH, 64, 0, stream>>>(convout, dtb, dAb, D_param, ybuf);

    // gated RMSNorm -> bf16
    k_rmsnorm<<<ROWS, 256, 0, stream>>>(proj, ybuf, norm_w, yn_bf);

    // GEMM2: out = yn @ W_out        (8192 x 2048 x 2048)
    {
        dim3 grid(DM / 128, ROWS / 128);
        k_gemm_bf16<<<grid, 128, 0, stream>>>(yn_bf, WoutT_bf, (const float*)nullptr, out, ROWS, DM, DM);
    }
}
